// SkelPointEdgeNet_85667417686369
// MI455X (gfx1250) — matrix operand order repaired and checked by reading, not compiled, not run on hardware
//
#include <hip/hip_runtime.h>

// ---------------------------------------------------------------------------
// SkelPointEdgeNet for MI455X (gfx1250).
// GEMMs: 256-thread blocks (8 waves), block tile 128M x 64N.
//   - B (weights, pre-transposed bf16 [Co,Kp]) staged 64x32 per k-step into
//     LDS, DOUBLE-BUFFERED and software pipelined: next chunk is fetched
//     (GLOBAL_LOAD_ASYNC_TO_LDS_B128 when the builtin exists, else
//     global->reg->ds_store) while the current chunk feeds 4 independent
//     v_wmma_f32_16x16x32_bf16 accumulator chains per wave.  One barrier
//     per k-step.
//   - A fragment for k+32 prefetched into registers during compute.
// EdgeConv fuses max-over-k into the WMMA epilogue via monotonic
// float->uint atomicMax keys; BN stats (sum/sumsq) accumulate atomically.
// ---------------------------------------------------------------------------

typedef __attribute__((ext_vector_type(16))) __bf16 v16bf;
typedef __attribute__((ext_vector_type(8)))  float  v8f;
typedef __attribute__((ext_vector_type(8)))  unsigned short us8;
typedef __attribute__((ext_vector_type(4)))  int    v4i;

#define STATS_STRIDE 2048

#if __has_builtin(__builtin_amdgcn_global_load_async_to_lds_b128) && \
    __has_builtin(__builtin_amdgcn_s_wait_asynccnt)
#define HAVE_ASYNC_LDS 1
#else
#define HAVE_ASYNC_LDS 0
#endif

__device__ __forceinline__ unsigned short f2bf(float f) {
  unsigned u = __float_as_uint(f);
  unsigned r = u + 0x7FFFu + ((u >> 16) & 1u);   // round-to-nearest-even
  return (unsigned short)(r >> 16);
}
__device__ __forceinline__ unsigned fkey(float f) {
  unsigned u = __float_as_uint(f);
  return (u & 0x80000000u) ? ~u : (u | 0x80000000u);   // monotonic order key
}
__device__ __forceinline__ float funkey(unsigned k) {
  unsigned u = (k & 0x80000000u) ? (k & 0x7FFFFFFFu) : ~k;
  return __uint_as_float(u);
}

#if HAVE_ASYNC_LDS
// 16B async copy global -> LDS (tracked by ASYNCcnt).  Builtin signature
// (from the round-3 diagnostic): (int4 addrspace(1)*, int4 addrspace(3)*,
// imm offset, imm cpol).  LDS offset = low 32 bits of the generic address
// (ISA aperture model).
typedef __attribute__((address_space(1))) v4i gv4i;
typedef __attribute__((address_space(3))) v4i lv4i;
__device__ __forceinline__ void stage_async(unsigned short* ldst,
                                            const unsigned short* gsrc) {
  __builtin_amdgcn_global_load_async_to_lds_b128(
      (gv4i*)(uintptr_t)gsrc,
      (lv4i*)(unsigned)(uintptr_t)ldst,
      0, 0);
}
#endif

// ---------------------------------------------------------------------------
// WMMA GEMM.  A: [Mp, Kp] bf16 row-major (Mp % 128 == 0, rows >= Mv zeroed).
// Bt: [Np, Kp] bf16 row-major (B transposed, Np % 64 == 0, rows >= Nc zeroed).
// mode 0: Y[row*ldy + n] = acc  (row < Mv)
// mode 1: atomicMax(MB[(row/Kn)*Nc + n], key(acc))   (EdgeConv max over k)
// Both modes accumulate per-channel sum/sumsq into stats (BatchNorm); zero
// padded rows contribute exactly 0 to both sums.
// ---------------------------------------------------------------------------
__global__ __launch_bounds__(256)
void k_gemm_bf16(const unsigned short* __restrict__ A,
                 const unsigned short* __restrict__ Bt,
                 float* __restrict__ Y, unsigned* __restrict__ MB,
                 float* __restrict__ stats,
                 int Kp, int Nc, int ldy, int Kn, int Mv, int mode) {
  __shared__ unsigned short lb[2][64 * 32];         // 2 x (64 cols x 32 K)
  const int tid  = threadIdx.x;
  const int wave = tid >> 5;
  const int lane = tid & 31;
  const int half = lane >> 4;
  const int l16  = lane & 15;
  const int m0   = blockIdx.x * 128 + wave * 16;
  const int n0   = blockIdx.y * 64;

  v8f acc[4] = {};
  const unsigned short* arow = A + (size_t)(m0 + l16) * Kp;
  // cooperative B loader: thread t -> row t/4 (of 64), cols (t%4)*8..+7
  const int brow_ = tid >> 2;
  const int bcol_ = (tid & 3) * 8;
  const int lofs  = brow_ * 32 + bcol_;
  const unsigned short* bsrc = Bt + (size_t)(n0 + brow_) * Kp + bcol_;

  union F { us8 h[2]; v16bf v; };

  // ---- prologue: stage k=0 chunk into buffer 0 ----------------------------
#if HAVE_ASYNC_LDS
  stage_async(&lb[0][lofs], bsrc);
  __builtin_amdgcn_s_wait_asynccnt(0);
#else
  *(us8*)(&lb[0][lofs]) = *(const us8*)(bsrc);
#endif
  __syncthreads();

  // A frag for k=0: lane<16 -> M=l16, K {0..7, 16..23}; lane>=16 -> +8
  F afc;
  afc.h[0] = *(const us8*)(arow + half * 8);
  afc.h[1] = *(const us8*)(arow + 16 + half * 8);

  int buf = 0;
  for (int k = 0; k < Kp; k += 32, buf ^= 1) {
    const bool hasNext = (k + 32) < Kp;
    // ---- stage k+32 chunk into the other buffer (overlaps compute) --------
#if HAVE_ASYNC_LDS
    if (hasNext) stage_async(&lb[buf ^ 1][lofs], bsrc + k + 32);
#else
    us8 bnext;
    if (hasNext) bnext = *(const us8*)(bsrc + k + 32);
#endif
    // ---- prefetch next A fragment into registers --------------------------
    F afn;
    if (hasNext) {
      afn.h[0] = *(const us8*)(arow + k + 32 + half * 8);
      afn.h[1] = *(const us8*)(arow + k + 48 + half * 8);
    }
    // ---- B 32x16 frags from LDS: lane = column, K {0..7,16..23} (+8 upper half)
    const unsigned short* lbase = &lb[buf][l16 * 32 + half * 8];
    F bf0, bf1, bf2, bf3;
    bf0.h[0] = *(const us8*)(lbase);          bf0.h[1] = *(const us8*)(lbase + 16);
    bf1.h[0] = *(const us8*)(lbase + 512);    bf1.h[1] = *(const us8*)(lbase + 528);
    bf2.h[0] = *(const us8*)(lbase + 1024);   bf2.h[1] = *(const us8*)(lbase + 1040);
    bf3.h[0] = *(const us8*)(lbase + 1536);   bf3.h[1] = *(const us8*)(lbase + 1552);
    acc[0] = __builtin_amdgcn_wmma_f32_16x16x32_bf16(
        false, afc.v, false, bf0.v, (short)0, acc[0], false, false);
    acc[1] = __builtin_amdgcn_wmma_f32_16x16x32_bf16(
        false, afc.v, false, bf1.v, (short)0, acc[1], false, false);
    acc[2] = __builtin_amdgcn_wmma_f32_16x16x32_bf16(
        false, afc.v, false, bf2.v, (short)0, acc[2], false, false);
    acc[3] = __builtin_amdgcn_wmma_f32_16x16x32_bf16(
        false, afc.v, false, bf3.v, (short)0, acc[3], false, false);
#if !HAVE_ASYNC_LDS
    if (hasNext) *(us8*)(&lb[buf ^ 1][lofs]) = bnext;
#endif
#if HAVE_ASYNC_LDS
    __builtin_amdgcn_s_wait_asynccnt(0);
#endif
    __syncthreads();
    if (hasNext) afc = afn;
  }

  const int mbase = m0 + half * 8;     // C/D layout: VGPR r -> M = mbase + r
#pragma unroll
  for (int j = 0; j < 4; ++j) {
    const int n = n0 + j * 16 + l16;
    if (n < Nc) {
      float s = 0.f, ss = 0.f;
#pragma unroll
      for (int r = 0; r < 8; ++r) { float v = acc[j][r]; s += v; ss += v * v; }
      atomicAdd(&stats[n], s);
      atomicAdd(&stats[STATS_STRIDE + n], ss);
      if (mode == 0) {
#pragma unroll
        for (int r = 0; r < 8; ++r) {
          int row = mbase + r;
          if (row < Mv) Y[(size_t)row * ldy + n] = acc[j][r];
        }
      } else {
#pragma unroll
        for (int r = 0; r < 8; ++r) {
          int row = mbase + r;
          atomicMax(&MB[(size_t)(row / Kn) * Nc + n], fkey(acc[j][r]));
        }
      }
    }
  }
}

// f32 -> bf16 with zero padding of K to Kp and rows to Mp
__global__ __launch_bounds__(256)
void k_cvt_pad(const float* __restrict__ src, int lds,
               unsigned short* __restrict__ dst, int Kp, int C, int Mrows,
               int total) {
  int t = blockIdx.x * 256 + threadIdx.x;
  if (t >= total) return;
  int row = t / Kp, k = t - row * Kp;
  dst[t] = (k < C && row < Mrows) ? f2bf(src[(size_t)row * lds + k])
                                  : (unsigned short)0;
}

// Weight transpose + convert:  Wt[n*Kp+k] = bf16(W[k*Co+n]), zero-padded
__global__ __launch_bounds__(256)
void k_wt(const float* __restrict__ W, int Cin, int Co,
          unsigned short* __restrict__ Wt, int Kp, int total) {
  int t = blockIdx.x * 256 + threadIdx.x;
  if (t >= total) return;
  int n = t / Kp, k = t - n * Kp;
  Wt[t] = (k < Cin && n < Co) ? f2bf(W[(size_t)k * Co + n]) : (unsigned short)0;
}

// Brute-force KNN: one thread per query, insertion-sorted top-kk.
// Tie-break (strict compares, ascending candidate order) matches lax.top_k.
__global__ __launch_bounds__(64)
void k_knn(const float* __restrict__ X, int ldx,
           const float* __restrict__ Y, int ldy,
           int C, int P, int Q, int kk, int* __restrict__ out, int total) {
  int t = blockIdx.x * 64 + threadIdx.x;
  if (t >= total) return;
  int b = t / P, p = t - b * P;
  const float* xr = X + ((size_t)b * P + p) * ldx;
  const float* yb = Y + (size_t)b * Q * ldy;
  float bd[20]; int bi[20];
  for (int j = 0; j < kk; ++j) { bd[j] = 3.4e38f; bi[j] = 0; }
  for (int q = 0; q < Q; ++q) {
    const float* yr = yb + (size_t)q * ldy;
    __builtin_prefetch(yr + 16 * ldy, 0, 1);     // global_prefetch_b8
    float d = 0.f;
    for (int c = 0; c < C; ++c) { float df = xr[c] - yr[c]; d += df * df; }
    if (d < bd[kk - 1]) {
      int j = kk - 1;
      while (j > 0 && bd[j - 1] > d) { bd[j] = bd[j - 1]; bi[j] = bi[j - 1]; --j; }
      bd[j] = d; bi[j] = q;
    }
  }
  int* o = out + (size_t)t * kk;
  for (int j = 0; j < kk; ++j) o[j] = bi[j];
}

// EdgeConv feature build (one batch): gf[n*K+kq, :] = [nbr-x | x] -> bf16 pad
__global__ __launch_bounds__(256)
void k_build_gf(const float* __restrict__ X, int ldx, const int* __restrict__ idx,
                int K, int C, int Kp, unsigned short* __restrict__ gf, int total) {
  int t = blockIdx.x * 256 + threadIdx.x;
  if (t >= total) return;
  int row = t / Kp, kc = t - row * Kp;
  unsigned short v = 0;
  if (kc < 2 * C) {
    int n = row / K;
    float val;
    if (kc < C) {
      int nb = idx[row];
      val = X[(size_t)nb * ldx + kc] - X[(size_t)n * ldx + kc];
    } else {
      val = X[(size_t)n * ldx + (kc - C)];
    }
    v = f2bf(val);
  }
  gf[t] = v;
}

// BatchNorm(train stats) + (Leaky)ReLU apply.  MB!=null: decode max keys.
__global__ __launch_bounds__(256)
void k_bn_apply(const float* __restrict__ Y, const unsigned* __restrict__ MB,
                const float* __restrict__ stats,
                const float* __restrict__ gma, const float* __restrict__ bta,
                float invCnt, float slope,
                float* __restrict__ dst, int ldd, int colOff, int Co, int total) {
  int t = blockIdx.x * 256 + threadIdx.x;
  if (t >= total) return;
  int row = t / Co, c = t - row * Co;
  float f = MB ? funkey(MB[t]) : Y[t];
  float m = stats[c] * invCnt;
  float var = stats[STATS_STRIDE + c] * invCnt - m * m;
  float v = (f - m) * rsqrtf(var + 1e-5f) * gma[c] + bta[c];
  v = v > 0.f ? v : slope * v;
  dst[(size_t)row * ldd + colOff + c] = v;
}

// Global max pool over N: g[b,c] = max_n x5[b,n,c]
__global__ __launch_bounds__(256)
void k_gpool(const float* __restrict__ x5, float* __restrict__ g,
             int N, int C, int total) {
  int t = blockIdx.x * 256 + threadIdx.x;
  if (t >= total) return;
  int b = t / C, c = t - b * C;
  float m = -3.4e38f;
  for (int n = 0; n < N; ++n)
    m = fmaxf(m, x5[((size_t)b * N + n) * C + c]);
  g[t] = m;
}

// c6 input = [broadcast(g) | xcat]  -> bf16 [B*N, 1536]
__global__ __launch_bounds__(256)
void k_build_c6in(const float* __restrict__ g, const float* __restrict__ xcat,
                  unsigned short* __restrict__ dst, int N, int total) {
  int t = blockIdx.x * 256 + threadIdx.x;
  if (t >= total) return;
  int row = t / 1536, c = t - row * 1536;
  int b = row / N;
  float v = (c < 1024) ? g[(size_t)b * 1024 + c]
                       : xcat[(size_t)row * 512 + (c - 1024)];
  dst[t] = f2bf(v);
}

// Farthest point sampling, one block per batch.  N=2048, S=100 hardcoded.
__global__ __launch_bounds__(256)
void k_fps(const float* __restrict__ pc, int* __restrict__ fidx) {
  const int b = blockIdx.x, tid = threadIdx.x;
  __shared__ float sd[256];
  __shared__ int   si[256];
  __shared__ int   sfar;
  const float* P = pc + (size_t)b * 2048 * 3;
  float dloc[8];
#pragma unroll
  for (int j = 0; j < 8; ++j) dloc[j] = 1e10f;
  int far = 0;
  for (int s = 0; s < 100; ++s) {
    if (tid == 0) fidx[b * 100 + s] = far;
    float cx = P[far * 3], cy = P[far * 3 + 1], cz = P[far * 3 + 2];
    float lm = -1.f; int li = 0;
#pragma unroll
    for (int j = 0; j < 8; ++j) {
      int p = tid * 8 + j;                       // contiguous -> tie = low idx
      float dx = P[p * 3] - cx, dy = P[p * 3 + 1] - cy, dz = P[p * 3 + 2] - cz;
      float dd = dx * dx + dy * dy + dz * dz;
      dloc[j] = fminf(dloc[j], dd);
      if (dloc[j] > lm) { lm = dloc[j]; li = p; }
    }
    sd[tid] = lm; si[tid] = li;
    __syncthreads();
    for (int w = 128; w > 0; w >>= 1) {
      if (tid < w) {
        if (sd[tid + w] > sd[tid] ||
            (sd[tid + w] == sd[tid] && si[tid + w] < si[tid])) {
          sd[tid] = sd[tid + w]; si[tid] = si[tid + w];
        }
      }
      __syncthreads();
    }
    if (tid == 0) sfar = si[0];
    __syncthreads();
    far = sfar;
    __syncthreads();
  }
}

__global__ __launch_bounds__(256)
void k_gather_xyz(const float* __restrict__ pc, const int* __restrict__ fidx,
                  float* __restrict__ nxyz, int N, int S, int total) {
  int t = blockIdx.x * 256 + threadIdx.x;
  if (t >= total) return;
  int q = fidx[t];
  int b = t / S;
  for (int c = 0; c < 3; ++c)
    nxyz[(size_t)t * 3 + c] = pc[((size_t)b * N + q) * 3 + c];
}

// nf0 = [gxyz - new_xyz | gathered feats]  (f32 [3600, 515]); also save gxyz
__global__ __launch_bounds__(256)
void k_build_nf(const float* __restrict__ pc, const float* __restrict__ nxyz,
                const int* __restrict__ dly, const float* __restrict__ feats,
                float* __restrict__ nf, float* __restrict__ gxyz, int total) {
  int t = blockIdx.x * 256 + threadIdx.x;
  if (t >= total) return;
  int row = t / 515, c = t - row * 515;
  int b = row / 900, r2 = row - b * 900;
  int s = r2 / 9, j = r2 - s * 9;
  int q = dly[((size_t)b * 100 + s) * 9 + j];
  float v;
  if (c < 3) {
    float pv = pc[((size_t)b * 2048 + q) * 3 + c];
    gxyz[(size_t)row * 3 + c] = pv;
    v = pv - nxyz[((size_t)b * 100 + s) * 3 + c];
  } else {
    v = feats[((size_t)b * 2048 + q) * 512 + (c - 3)];
  }
  nf[t] = v;
}

// w = max_k nf[...,k,:]; skel_xyz = sum w*gxyz; skel_r = sum w*|gxyz-skel|
__global__ __launch_bounds__(64)
void k_final(const float* __restrict__ nf, const float* __restrict__ gxyz,
             float* __restrict__ out, int total) {
  int t = blockIdx.x * 64 + threadIdx.x;
  if (t >= total) return;
  float w[9];
  for (int c = 0; c < 9; ++c) {
    float m = -3.4e38f;
    for (int k = 0; k < 9; ++k)
      m = fmaxf(m, nf[((size_t)t * 9 + k) * 9 + c]);
    w[c] = m;
  }
  float sx = 0.f, sy = 0.f, sz = 0.f;
  for (int j = 0; j < 9; ++j) {
    const float* g = gxyz + ((size_t)t * 9 + j) * 3;
    sx += w[j] * g[0]; sy += w[j] * g[1]; sz += w[j] * g[2];
  }
  float sr = 0.f;
  for (int j = 0; j < 9; ++j) {
    const float* g = gxyz + ((size_t)t * 9 + j) * 3;
    float dx = g[0] - sx, dy = g[1] - sy, dz = g[2] - sz;
    sr += w[j] * sqrtf(dx * dx + dy * dy + dz * dz + 1e-12f);
  }
  out[t * 4 + 0] = sx; out[t * 4 + 1] = sy;
  out[t * 4 + 2] = sz; out[t * 4 + 3] = sr;
}

// ---------------------------------------------------------------------------
extern "C" void kernel_launch(void* const* d_in, const int* in_sizes, int n_in,
                              void* d_out, int out_size, void* d_ws, size_t ws_size,
                              hipStream_t stream) {
  (void)in_sizes; (void)n_in; (void)out_size; (void)ws_size;
  const int B = 4, N = 2048, K = 20, S = 100, EMB = 1024;

  const float* pc = (const float*)d_in[0];
  const float *cW[7], *cG[7], *cB[7];
  int p = 1;
  for (int i = 0; i < 7; ++i) {
    cW[i] = (const float*)d_in[p++];
    cG[i] = (const float*)d_in[p++];
    cB[i] = (const float*)d_in[p++];
  }
  const float *mW[5], *mG[5], *mB[5];
  for (int i = 0; i < 5; ++i) {
    mW[i] = (const float*)d_in[p++];
    p++;                                   // bias: cancels under BatchNorm
    mG[i] = (const float*)d_in[p++];
    mB[i] = (const float*)d_in[p++];
  }

  // workspace bump allocator (256B aligned)
  char* wsb = (char*)d_ws; size_t off = 0;
  auto alloc = [&](size_t bytes) -> void* {
    void* r = wsb + off; off = (off + bytes + 255) & ~(size_t)255; return r;
  };
  float*  xcat   = (float*)alloc((size_t)B * N * 512 * 4);
  float*  x5     = (float*)alloc((size_t)B * N * 1024 * 4);
  float*  featsA = (float*)alloc((size_t)B * N * 512 * 4);
  float*  featsB = (float*)alloc((size_t)B * N * 512 * 4);
  float*  ybuf   = (float*)alloc((size_t)B * N * 1024 * 4);
  unsigned short* abuf = (unsigned short*)alloc((size_t)N * K * 512 * 2); // 42MB
  unsigned short* wbuf = (unsigned short*)alloc((size_t)1024 * 1536 * 2);
  unsigned* mbuf = (unsigned*)alloc((size_t)B * N * 256 * 4);
  float*  stats  = (float*)alloc((size_t)2 * STATS_STRIDE * 4);
  int*    knnIdx = (int*)alloc((size_t)B * N * K * 4);
  float*  gpool  = (float*)alloc((size_t)B * EMB * 4);
  int*    fidx   = (int*)alloc((size_t)B * S * 4);
  float*  nxyz   = (float*)alloc((size_t)B * S * 3 * 4);
  int*    dly    = (int*)alloc((size_t)B * S * 9 * 4);
  float*  gxyz   = (float*)alloc((size_t)B * S * 9 * 3 * 4);
  float*  nf     = (float*)alloc((size_t)B * S * 9 * 544 * 4);

  auto cdiv = [](int a, int b) { return (a + b - 1) / b; };

  // ---------------- EdgeConv 1..4 (fused max-over-k in WMMA epilogue) ------
  struct ECfg { int Cin, Co, dstOff; const float* X; int ldx; int wi; };
  ECfg ec[4] = {
    {3,   64,  0,   pc,         3,   0},
    {64,  64,  64,  xcat + 0,   512, 1},
    {64,  128, 128, xcat + 64,  512, 2},
    {128, 256, 256, xcat + 128, 512, 3},
  };
  for (int L = 0; L < 4; ++L) {
    const int Cin = ec[L].Cin, Co = ec[L].Co;   // Co: 64/64/128/256 (all %64)
    const int Kd = 2 * Cin, Kp = ((Kd + 31) / 32) * 32;
    int tot = B * N;
    k_knn<<<cdiv(tot, 64), 64, 0, stream>>>(ec[L].X, ec[L].ldx, ec[L].X,
                                            ec[L].ldx, Cin, N, N, K, knnIdx, tot);
    (void)hipMemsetAsync(stats, 0, (size_t)2 * STATS_STRIDE * 4, stream);
    (void)hipMemsetAsync(mbuf, 0, (size_t)B * N * Co * 4, stream);
    int wt = Co * Kp;
    k_wt<<<cdiv(wt, 256), 256, 0, stream>>>(cW[ec[L].wi], Kd, Co, wbuf, Kp, wt);
    for (int b = 0; b < B; ++b) {
      const float* Xb = ec[L].X + (size_t)b * N * ec[L].ldx;
      const int* idxb = knnIdx + (size_t)b * N * K;
      int gt = N * K * Kp;
      k_build_gf<<<cdiv(gt, 256), 256, 0, stream>>>(Xb, ec[L].ldx, idxb,
                                                    K, Cin, Kp, abuf, gt);
      dim3 gg(N * K / 128, Co / 64);            // 40960 rows: %128 == 0
      k_gemm_bf16<<<gg, 256, 0, stream>>>(abuf, wbuf, nullptr,
                                          mbuf + (size_t)b * N * Co, stats,
                                          Kp, Co, Co, K, N * K, 1);
    }
    int at = B * N * Co;
    k_bn_apply<<<cdiv(at, 256), 256, 0, stream>>>(
        nullptr, mbuf, stats, cG[ec[L].wi], cB[ec[L].wi],
        1.f / ((float)B * N * K), 0.2f, xcat, 512, ec[L].dstOff, Co, at);
  }

  const int M = B * N;  // 8192 (%128 == 0)
  // ---------------- c5: xcat[8192,512] -> x5[8192,1024] --------------------
  {
    int ct = M * 512;
    k_cvt_pad<<<cdiv(ct, 256), 256, 0, stream>>>(xcat, 512, abuf, 512, 512, M, ct);
    (void)hipMemsetAsync(stats, 0, (size_t)2 * STATS_STRIDE * 4, stream);
    int wt = 1024 * 512;
    k_wt<<<cdiv(wt, 256), 256, 0, stream>>>(cW[4], 512, 1024, wbuf, 512, wt);
    dim3 gg(M / 128, 1024 / 64);
    k_gemm_bf16<<<gg, 256, 0, stream>>>(abuf, wbuf, ybuf, nullptr, stats,
                                        512, 1024, 1024, 0, M, 0);
    int at = M * 1024;
    k_bn_apply<<<cdiv(at, 256), 256, 0, stream>>>(
        ybuf, nullptr, stats, cG[4], cB[4], 1.f / (float)M, 0.2f,
        x5, 1024, 0, 1024, at);
  }
  // ---------------- global max pool + c6 + c7 ------------------------------
  {
    int gt = B * EMB;
    k_gpool<<<cdiv(gt, 256), 256, 0, stream>>>(x5, gpool, N, EMB, gt);
    int bt = M * 1536;
    k_build_c6in<<<cdiv(bt, 256), 256, 0, stream>>>(gpool, xcat, abuf, N, bt);
    (void)hipMemsetAsync(stats, 0, (size_t)2 * STATS_STRIDE * 4, stream);
    int wt = 512 * 1536;
    k_wt<<<cdiv(wt, 256), 256, 0, stream>>>(cW[5], 1536, 512, wbuf, 1536, wt);
    dim3 gg(M / 128, 512 / 64);
    k_gemm_bf16<<<gg, 256, 0, stream>>>(abuf, wbuf, ybuf, nullptr, stats,
                                        1536, 512, 512, 0, M, 0);
    int at = M * 512;
    k_bn_apply<<<cdiv(at, 256), 256, 0, stream>>>(
        ybuf, nullptr, stats, cG[5], cB[5], 1.f / (float)M, 0.2f,
        featsA, 512, 0, 512, at);

    int ct = M * 512;
    k_cvt_pad<<<cdiv(ct, 256), 256, 0, stream>>>(featsA, 512, abuf, 512, 512, M, ct);
    (void)hipMemsetAsync(stats, 0, (size_t)2 * STATS_STRIDE * 4, stream);
    wt = 512 * 512;
    k_wt<<<cdiv(wt, 256), 256, 0, stream>>>(cW[6], 512, 512, wbuf, 512, wt);
    k_gemm_bf16<<<gg, 256, 0, stream>>>(abuf, wbuf, ybuf, nullptr, stats,
                                        512, 512, 512, 0, M, 0);
    k_bn_apply<<<cdiv(at, 256), 256, 0, stream>>>(
        ybuf, nullptr, stats, cG[6], cB[6], 1.f / (float)M, 0.2f,
        featsB, 512, 0, 512, at);
  }
  // ---------------- FPS, skeleton KNN, nf build ----------------------------
  k_fps<<<B, 256, 0, stream>>>(pc, fidx);
  {
    int gt = B * S;
    k_gather_xyz<<<cdiv(gt, 256), 256, 0, stream>>>(pc, fidx, nxyz, N, S, gt);
    k_knn<<<cdiv(gt, 64), 64, 0, stream>>>(nxyz, 3, pc, 3, 3, S, N, 9, dly, gt);
    int nt = 3600 * 515;
    k_build_nf<<<cdiv(nt, 256), 256, 0, stream>>>(pc, nxyz, dly, featsB,
                                                  nf, gxyz, nt);
  }
  // ---------------- MLP: 515->384->256->256->128->9 ------------------------
  {
    const int mdims[6] = {515, 384, 256, 256, 128, 9};
    const int Mr = B * S * 9;                      // 3600 valid rows
    const int Mp = ((Mr + 127) / 128) * 128;       // 3712 padded (zero rows)
    for (int i = 0; i < 5; ++i) {
      int Kd = mdims[i], Co = mdims[i + 1];
      int Kp = ((Kd + 31) / 32) * 32;
      int Np = ((Co + 63) / 64) * 64;
      int ct = Mp * Kp;
      k_cvt_pad<<<cdiv(ct, 256), 256, 0, stream>>>(nf, Kd, abuf, Kp, Kd, Mr, ct);
      (void)hipMemsetAsync(stats, 0, (size_t)2 * STATS_STRIDE * 4, stream);
      int wt = Np * Kp;
      k_wt<<<cdiv(wt, 256), 256, 0, stream>>>(mW[i], Kd, Co, wbuf, Kp, wt);
      dim3 gg(Mp / 128, Np / 64);
      k_gemm_bf16<<<gg, 256, 0, stream>>>(abuf, wbuf, ybuf, nullptr, stats,
                                          Kp, Co, Co, 0, Mr, 0);
      int at = Mr * Co;
      k_bn_apply<<<cdiv(at, 256), 256, 0, stream>>>(
          ybuf, nullptr, stats, mG[i], mB[i], 1.f / (float)Mr, 0.f,
          nf, Co, 0, Co, at);
    }
  }
  // ---------------- final weighted skeleton points -------------------------
  k_final<<<cdiv(B * S, 64), 64, 0, stream>>>(nf, gxyz, (float*)d_out, B * S);
}
